// MultiHeadedAttention_55362128445840
// MI455X (gfx1250) — matrix operand order repaired and checked by reading, not compiled, not run on hardware
//
#include <hip/hip_runtime.h>

// ---------- types ----------
typedef __attribute__((ext_vector_type(16))) __bf16 v16bf;
typedef __attribute__((ext_vector_type(8)))  float  v8f;
typedef unsigned short u16;
typedef __attribute__((ext_vector_type(16))) u16 v16u;
typedef __attribute__((ext_vector_type(8)))  u16 v8u;

#define B_SZ    8
#define NSEQ    2048
#define DMODEL  256
#define NHEAD   4
#define HDIM    64

// ---------- helpers ----------
__device__ __forceinline__ u16 f2bf(float f) {
    return __builtin_bit_cast(u16, (__bf16)f);     // hardware f32->bf16 (RNE)
}

__device__ __forceinline__ v16bf as_bf16x16(v16u u) {
    union { v16u u; v16bf b; } c; c.u = u; return c.b;
}

__device__ __forceinline__ v16u cat8(v8u lo, v8u hi) {
    return __builtin_shufflevector(lo, hi, 0,1,2,3,4,5,6,7,8,9,10,11,12,13,14,15);
}

__device__ __forceinline__ v8f wmma_bf16(v16bf a, v16bf b, v8f c) {
    // D = A(16x32 bf16) * B(32x16 bf16) + C(16x16 f32)
    return __builtin_amdgcn_wmma_f32_16x16x32_bf16(false, a, false, b,
                                                   (short)0, c, false, false);
}

// 16-bit A-operand: per lane (row = l16), one contiguous 16-elem run along K:
//   K = half*16 + [0..15]   (within a 32-wide K step)
__device__ __forceinline__ v16bf load_a_tile(const u16* rowptr, int half) {
    const u16* p = rowptr + half * 16;
    v8u lo = *(const v8u*)(p);
    v8u hi = *(const v8u*)(p + 8);
    return as_bf16x16(cat8(lo, hi));
}

// CDNA5 async copy: 16B global -> LDS, tracked with ASYNCcnt
__device__ __forceinline__ void async_b128(const u16* gsrc, u16* lds_dst) {
    unsigned l = (unsigned)(unsigned long long)(const void*)lds_dst;
    asm volatile("global_load_async_to_lds_b128 %0, %1, off"
                 :: "v"(l), "v"(gsrc) : "memory");
}

template <int N>
__device__ __forceinline__ void wait_async() {
    asm volatile("s_wait_asynccnt %0" :: "i"(N) : "memory");
}

// ---------- kernel 0a: f32 -> bf16 weight convert ----------
__global__ void cvt_w_kernel(const float* __restrict__ src, u16* __restrict__ dst, int n) {
    int i = blockIdx.x * 256 + threadIdx.x;
    if (i < n) dst[i] = f2bf(src[i]);
}

// ---------- kernel 0b: [B,C,N] f32 -> [B,N,C] bf16 transpose ----------
__global__ __launch_bounds__(256) void transpose_cvt_kernel(const float* __restrict__ in,
                                                            u16* __restrict__ out) {
    __shared__ float t[32][33];
    const int b  = blockIdx.z;
    const int n0 = blockIdx.x * 32;
    const int c0 = blockIdx.y * 32;
    const float* ib = in  + (size_t)b * DMODEL * NSEQ;
    u16*         ob = out + (size_t)b * NSEQ * DMODEL;
    #pragma unroll
    for (int i = 0; i < 32; i += 8)
        t[threadIdx.y + i][threadIdx.x] =
            ib[(size_t)(c0 + threadIdx.y + i) * NSEQ + n0 + threadIdx.x];
    __syncthreads();
    #pragma unroll
    for (int i = 0; i < 32; i += 8)
        ob[(size_t)(n0 + threadIdx.y + i) * DMODEL + c0 + threadIdx.x] =
            f2bf(t[threadIdx.x][threadIdx.y + i]);
}

// ---------- GEMM: Y[o,n] = sum_c W[o,c] * X[n,c] + bias[o] ----------
// X panel (64 rows x 256) staged once into LDS with async copies (shared by all 4 waves).
// MODE 0: bf16 Q/K layout [B,H,N,D];  MODE 1: bf16 V layout [B,H,D,N];  MODE 2: f32 [B,C,N].
template <int MODE>
__global__ __launch_bounds__(128) void proj_gemm_kernel(const u16* __restrict__ Wb,
                                                        const float* __restrict__ bias,
                                                        const u16* __restrict__ X,
                                                        void* __restrict__ out) {
    __shared__ alignas(32) u16 xs[64 * DMODEL];    // 32 KB

    const int tid   = threadIdx.x;
    const int lane  = tid & 31;
    const int wave  = tid >> 5;
    const int half  = lane >> 4;
    const int l16   = lane & 15;
    const int b     = blockIdx.z;
    const int nbase = blockIdx.x * 64;
    const int obase = blockIdx.y * 64 + wave * 16;

    // stage X[nbase..nbase+64) x 256 : 32 KB contiguous -> LDS
    const u16* Xp = X + (size_t)b * NSEQ * DMODEL + (size_t)nbase * DMODEL;
    #pragma unroll
    for (int i = 0; i < 16; i++) {
        int c = tid + i * 128;                     // 16B chunk id, 0..2047
        async_b128(Xp + c * 8, &xs[c * 8]);
    }
    wait_async<0>();
    __syncthreads();

    v8f acc[4] = {};
    #pragma unroll
    for (int ks = 0; ks < DMODEL; ks += 32) {
        v16bf av = load_a_tile(Wb + (size_t)(obase + l16) * DMODEL + ks, half);
        #pragma unroll
        for (int t = 0; t < 4; t++) {
            const u16* bp = &xs[(t * 16 + l16) * DMODEL + ks + half * 16];
            acc[t] = wmma_bf16(av, as_bf16x16(*(const v16u*)bp), acc[t]);
        }
    }

    #pragma unroll
    for (int t = 0; t < 4; t++) {
        const int n = nbase + t * 16 + l16;
        #pragma unroll
        for (int r = 0; r < 8; r++) {
            const int o = obase + r + 8 * half;     // C/D layout: row = r + 8*half
            float v = acc[t][r] + bias[o];
            if constexpr (MODE == 2) {
                ((float*)out)[((size_t)b * DMODEL + o) * NSEQ + n] = v;
            } else {
                const int hh = o & 3, d = o >> 2;   // reshape(HEAD_DIM, NUM_HEADS)
                size_t idx = (MODE == 0)
                    ? ((((size_t)b * NHEAD + hh) * NSEQ + n) * HDIM + d)    // [B,H,N,D]
                    : ((((size_t)b * NHEAD + hh) * HDIM + d) * NSEQ + n);   // [B,H,D,N]
                ((u16*)out)[idx] = f2bf(v);
            }
        }
    }
}

// ---------- flash attention per (b,h): O = softmax(Q K^T / 8) V ----------
// Q,K: [B,H,N,D] bf16.  V: [B,H,D,N] bf16.  Out: [B,N,C] bf16 (c = d*4 + h).
// Scores computed TRANSPOSED (S^T = K Q^T): the S^T C/D register layout coincides
// lane-for-lane with the P A-operand layout, so P goes straight from registers into
// the P*V WMMA with no LDS round-trip. Softmax rows (over m) live in-lane: 32 values
// per lane + one shfl_xor(16) merge. K/V tiles double-buffered in LDS via async copies.
__global__ __launch_bounds__(128) void flash_attn_kernel(const u16* __restrict__ Q,
                                                         const u16* __restrict__ K,
                                                         const u16* __restrict__ V,
                                                         u16* __restrict__ O) {
    __shared__ alignas(32) u16 kbuf[2][64 * 64];    // 2 x 8 KB  [m_local][d]
    __shared__ alignas(32) u16 vbuf[2][64 * 64];    // 2 x 8 KB  [d][m_local]

    const int tid   = threadIdx.x;
    const int lane  = tid & 31;
    const int wave  = tid >> 5;
    const int half  = lane >> 4;
    const int l16   = lane & 15;
    const int bh    = blockIdx.y;                   // b*NHEAD + head
    const int b     = bh >> 2, head = bh & 3;
    const int nbase = blockIdx.x * 64 + wave * 16;

    const size_t off = (size_t)bh * NSEQ * HDIM;
    const u16* Qh = Q + off;
    const u16* Kh = K + off;
    const u16* Vh = V + off;

    // all-ones bf16 B-operand (for row sums of P via WMMA)
    union { u16 a[16]; v16u v; } uo;
    #pragma unroll
    for (int j = 0; j < 16; j++) uo.a[j] = 0x3F80;  // bf16(1.0)
    const v16bf vones = as_bf16x16(uo.v);

    // Q as two B-operands (d in [0,32) and [32,64)); col n = nbase + l16, contiguous d
    v16bf qB[2];
    #pragma unroll
    for (int ds = 0; ds < 2; ds++) {
        const u16* qp = Qh + (size_t)(nbase + l16) * HDIM + ds * 32 + half * 16;
        qB[ds] = as_bf16x16(*(const v16u*)qp);
    }

    v8f   oacc[4] = {};
    float l_r[8];                                   // row sums, O-layout (row = r + 8*half)
    #pragma unroll
    for (int r = 0; r < 8; r++) l_r[r] = 0.f;
    float m_n = -1e30f;                             // running max for row n = l16 (lane layout)

    const float c = 0.125f * 1.44269504088896f;     // 1/sqrt(64) * log2(e)

    // issue one 64-wide K/V tile into LDS buffer ib (8 async b128 per thread)
    auto issue_tile = [&](int mt, int ib) {
        #pragma unroll
        for (int i = 0; i < 4; i++) {               // K tile: 8 KB contiguous
            int ck = tid + i * 128;                 // 16B chunk 0..511
            async_b128(Kh + (size_t)mt * HDIM + ck * 8, &kbuf[ib][ck * 8]);
        }
        #pragma unroll
        for (int i = 0; i < 4; i++) {               // V tile: 64 rows x 128B
            int cv  = tid + i * 128;
            int row = cv >> 3, cc = cv & 7;
            async_b128(Vh + (size_t)row * NSEQ + mt + cc * 8,
                       &vbuf[ib][row * 64 + cc * 8]);
        }
    };

    issue_tile(0, 0);
    int ib = 0;
    for (int mt = 0; mt < NSEQ; mt += 64, ib ^= 1) {
        const bool more = (mt + 64 < NSEQ);
        if (more) { issue_tile(mt + 64, ib ^ 1); wait_async<8>(); }
        else      { wait_async<0>(); }
        __syncthreads();

        // ---- S^T = K * Q^T : tile t holds rows m = mt + t*16 + (r + 8*half), col n = l16
        v8f sT[4] = {};
        #pragma unroll
        for (int ds = 0; ds < 2; ds++) {
            #pragma unroll
            for (int t = 0; t < 4; t++) {
                v16bf ka = load_a_tile(&kbuf[ib][(t * 16 + l16) * 64 + ds * 32], half);
                sT[t] = wmma_bf16(ka, qB[ds], sT[t]);
            }
        }

        // ---- online softmax over m (in-lane + one cross-half merge) ----
        float mx = sT[0][0];
        #pragma unroll
        for (int t = 0; t < 4; t++)
            #pragma unroll
            for (int r = 0; r < 8; r++) mx = fmaxf(mx, sT[t][r]);
        mx *= c;
        mx = fmaxf(mx, __shfl_xor(mx, 16, 32));     // merge the two m-halves of column n
        const float nm    = fmaxf(m_n, mx);
        const float alpha = __builtin_amdgcn_exp2f(m_n - nm);
        m_n = nm;

        // broadcast alpha from lane-layout (n = l16) to O-layout rows (n = r + 8*half)
        float alpha_o[8];
        #pragma unroll
        for (int r = 0; r < 8; r++)
            alpha_o[r] = __shfl(alpha, r + 8 * half, 32);
        #pragma unroll
        for (int t2 = 0; t2 < 4; t2++)
            #pragma unroll
            for (int r = 0; r < 8; r++) oacc[t2][r] *= alpha_o[r];

        // ---- P in A-operand layout, straight from S^T registers ----
        // A element j of lane (l16,half): m = mc*32 + (j>=8)*16 + half*8 + (j&7)
        //                               = S^T tile [2*mc + (j>>3)], reg (j&7), same lane.
        v8f lsum = {};
        #pragma unroll
        for (int mc = 0; mc < 2; mc++) {
            v16bf pa;
            #pragma unroll
            for (int j = 0; j < 16; j++) {
                float p = __builtin_amdgcn_exp2f(sT[2 * mc + (j >> 3)][j & 7] * c - nm);
                pa[j] = (__bf16)p;
            }
            #pragma unroll
            for (int t2 = 0; t2 < 4; t2++) {
                const u16* vp = &vbuf[ib][(t2 * 16 + l16) * 64 + mc * 32 + half * 8];
                oacc[t2] = wmma_bf16(pa, as_bf16x16(cat8(*(const v8u*)vp, *(const v8u*)(vp + 16))), oacc[t2]);
            }
            lsum = wmma_bf16(pa, vones, lsum);      // row sums, already O-layout
        }
        #pragma unroll
        for (int r = 0; r < 8; r++) l_r[r] = l_r[r] * alpha_o[r] + lsum[r];

        __syncthreads();   // protect LDS buffer reuse across iterations
    }

    // ---- normalize and write [B,N,C] bf16, c = d*4 + head ----
    float inv[8];
    #pragma unroll
    for (int r = 0; r < 8; r++) inv[r] = __builtin_amdgcn_rcpf(l_r[r]);
    #pragma unroll
    for (int t2 = 0; t2 < 4; t2++) {
        const int d = t2 * 16 + l16;
        #pragma unroll
        for (int r = 0; r < 8; r++) {
            const int n = nbase + r + 8 * half;
            O[((size_t)b * NSEQ + n) * DMODEL + d * 4 + head] = f2bf(oacc[t2][r] * inv[r]);
        }
    }
}

// ---------- host ----------
extern "C" void kernel_launch(void* const* d_in, const int* in_sizes, int n_in,
                              void* d_out, int out_size, void* d_ws, size_t ws_size,
                              hipStream_t stream) {
    (void)in_sizes; (void)n_in; (void)out_size; (void)ws_size;
    const float* query = (const float*)d_in[0];
    const float* key_i = (const float*)d_in[1];
    const float* value = (const float*)d_in[2];
    const float* Wq = (const float*)d_in[3];  const float* bq = (const float*)d_in[4];
    const float* Wk = (const float*)d_in[5];  const float* bk = (const float*)d_in[6];
    const float* Wv = (const float*)d_in[7];  const float* bv = (const float*)d_in[8];
    const float* Wm = (const float*)d_in[9];  const float* bm = (const float*)d_in[10];

    char* ws = (char*)d_ws;
    const size_t WBYTES = (size_t)DMODEL * DMODEL * sizeof(u16);        // 128 KB
    const size_t XBYTES = (size_t)B_SZ * NSEQ * DMODEL * sizeof(u16);   // 8 MB
    u16* wq  = (u16*)(ws + 0 * WBYTES);
    u16* wk  = (u16*)(ws + 1 * WBYTES);
    u16* wv  = (u16*)(ws + 2 * WBYTES);
    u16* wm  = (u16*)(ws + 3 * WBYTES);
    char* base = ws + 4 * WBYTES;
    u16* xTq = (u16*)(base + 0 * XBYTES);
    u16* xTk = (u16*)(base + 1 * XBYTES);
    u16* xTv = (u16*)(base + 2 * XBYTES);
    u16* qb  = (u16*)(base + 3 * XBYTES);
    u16* kb_ = (u16*)(base + 4 * XBYTES);
    u16* vb  = xTq;   // reuse: xTq dead after Q projection
    u16* xws = xTk;   // reuse: xTk dead after K projection

    const int NW = DMODEL * DMODEL;
    cvt_w_kernel<<<NW / 256, 256, 0, stream>>>(Wq, wq, NW);
    cvt_w_kernel<<<NW / 256, 256, 0, stream>>>(Wk, wk, NW);
    cvt_w_kernel<<<NW / 256, 256, 0, stream>>>(Wv, wv, NW);
    cvt_w_kernel<<<NW / 256, 256, 0, stream>>>(Wm, wm, NW);

    dim3 tg(NSEQ / 32, DMODEL / 32, B_SZ), tb(32, 8, 1);
    transpose_cvt_kernel<<<tg, tb, 0, stream>>>(query, xTq);
    transpose_cvt_kernel<<<tg, tb, 0, stream>>>(key_i, xTk);
    transpose_cvt_kernel<<<tg, tb, 0, stream>>>(value, xTv);

    dim3 gg(NSEQ / 64, DMODEL / 64, B_SZ);
    proj_gemm_kernel<0><<<gg, 128, 0, stream>>>(wq, bq, xTq, (void*)qb);
    proj_gemm_kernel<0><<<gg, 128, 0, stream>>>(wk, bk, xTk, (void*)kb_);
    proj_gemm_kernel<1><<<gg, 128, 0, stream>>>(wv, bv, xTv, (void*)vb);

    dim3 ag(NSEQ / 64, B_SZ * NHEAD, 1);
    flash_attn_kernel<<<ag, 128, 0, stream>>>(qb, kb_, vb, xws);

    proj_gemm_kernel<2><<<gg, 128, 0, stream>>>(wm, bm, xws, d_out);
}
